// RetrainUtils_35107062677556
// MI455X (gfx1250) — matrix operand order repaired and checked by reading, not compiled, not run on hardware
//
#include <hip/hip_runtime.h>
#include <hip/hip_bf16.h>
#include <math.h>

typedef __attribute__((ext_vector_type(16))) _Float16 v16h;
typedef __attribute__((ext_vector_type(8)))  float    v8f;

#define NB   32
#define NA   8400
#define NG   50
#define NCLS 80
#define KPAD 96
#define ATILES 525   /* 8400 / 16 */

// ---------------------------------------------------------------- helpers
__device__ __forceinline__ void anchor_info(int n, int& gx, int& gy, float& stride,
                                            int& lvl, int& hw, int& loc) {
  if (n < 6400)      { lvl = 0; hw = 6400; int m = n;        gy = m / 80; gx = m - gy * 80; stride =  8.f; loc = m; }
  else if (n < 8000) { lvl = 1; hw = 1600; int m = n - 6400; gy = m / 40; gx = m - gy * 40; stride = 16.f; loc = m; }
  else               { lvl = 2; hw =  400; int m = n - 8000; gy = m / 20; gx = m - gy * 20; stride = 32.f; loc = m; }
}

__device__ __forceinline__ float read_raw(const float* o0, const float* o1, const float* o2,
                                          int b, int c, int n) {
  int gx, gy, lvl, hw, loc; float s;
  anchor_info(n, gx, gy, s, lvl, hw, loc);
  const float* p = (lvl == 0) ? o0 : ((lvl == 1) ? o1 : o2);
  return p[((size_t)b * 85 + c) * hw + loc];
}

__device__ __forceinline__ float sigm(float x) { return 1.f / (1.f + expf(-x)); }

__device__ __forceinline__ float bce_logit(float x, float y) {
  return fmaxf(x, 0.f) - x * y + log1pf(expf(-fabsf(x)));
}

// keep 10 largest, arr sorted descending
__device__ __forceinline__ void ins_desc(float* arr, float v) {
  if (v <= arr[9]) return;
  int i = 9;
  while (i > 0 && arr[i - 1] < v) { arr[i] = arr[i - 1]; --i; }
  arr[i] = v;
}
// keep 10 smallest, arr sorted ascending
__device__ __forceinline__ void ins_asc(float* arr, float v) {
  if (v >= arr[9]) return;
  int i = 9;
  while (i > 0 && arr[i - 1] > v) { arr[i] = arr[i - 1]; --i; }
  arr[i] = v;
}

// gt-side metadata bundle used by the WMMA epilogue
struct GtInfo {
  int   valid;
  float bx, by, bw, bh;
};

// ---------------------------------------------------------------- K0: GT metadata
// meta[b][g][6] = {valid, cls, cx, cy, w, h}
__global__ void k0_meta(const float* __restrict__ labels, float* __restrict__ meta) {
  int idx = blockIdx.x * blockDim.x + threadIdx.x;
  if (idx >= NB * NG) return;
  const float* l = labels + (size_t)idx * 5;
  float ssum = l[0] + l[1] + l[2] + l[3] + l[4];
  float* m = meta + (size_t)idx * 6;
  m[0] = (ssum > 0.f) ? 1.f : 0.f;
  m[1] = l[0]; m[2] = l[1]; m[3] = l[2]; m[4] = l[3]; m[5] = l[4];
}

// ---------------------------------------------------------------- K1: decode
__global__ void k1_decode(const float* __restrict__ o0, const float* __restrict__ o1,
                          const float* __restrict__ o2,
                          float* __restrict__ bbox, float* __restrict__ objL) {
  int idx = blockIdx.x * blockDim.x + threadIdx.x;
  if (idx >= NB * NA) return;
  int b = idx / NA, n = idx - b * NA;
  int gx, gy, lvl, hw, loc; float s;
  anchor_info(n, gx, gy, s, lvl, hw, loc);
  float tx = read_raw(o0, o1, o2, b, 0, n);
  float ty = read_raw(o0, o1, o2, b, 1, n);
  float tw = read_raw(o0, o1, o2, b, 2, n);
  float th = read_raw(o0, o1, o2, b, 3, n);
  float to = read_raw(o0, o1, o2, b, 4, n);
  float* pb = bbox + (size_t)idx * 4;
  pb[0] = (tx + (float)gx) * s;
  pb[1] = (ty + (float)gy) * s;
  pb[2] = expf(tw) * s;
  pb[3] = expf(th) * s;
  objL[idx] = to;
}

// ---------------------------------------------------------------- K2: Q = -logp + log1mp (f16), s = sum log1mp
__global__ void k2_prob(const float* __restrict__ o0, const float* __restrict__ o1,
                        const float* __restrict__ o2, const float* __restrict__ objL,
                        _Float16* __restrict__ Q, float* __restrict__ sArr) {
  int idx = blockIdx.x * blockDim.x + threadIdx.x;
  if (idx >= NB * NA) return;
  int b = idx / NA, n = idx - b * NA;
  float so = sigm(objL[idx]);
  float ssum = 0.f;
  _Float16* qrow = Q + (size_t)idx * KPAD;
  for (int c = 0; c < NCLS; ++c) {
    float x = read_raw(o0, o1, o2, b, 5 + c, n);
    float p = sqrtf(sigm(x) * so);
    p = fminf(fmaxf(p, 1e-7f), 1.f - 1e-6f);
    float lp = logf(p);
    float l1 = log1pf(-p);
    ssum += l1;
    qrow[c] = (_Float16)(l1 - lp);
  }
  for (int c = NCLS; c < KPAD; ++c) qrow[c] = (_Float16)0.0f;
  sArr[idx] = ssum;
}

// ---------------------------------------------------------------- K3: candidate-anchor mask fg
__global__ void k3_fg(const float* __restrict__ meta, int* __restrict__ fgC) {
  int idx = blockIdx.x * blockDim.x + threadIdx.x;
  if (idx >= NB * NA) return;
  int b = idx / NA, n = idx - b * NA;
  int gx, gy, lvl, hw, loc; float s;
  anchor_info(n, gx, gy, s, lvl, hw, loc);
  float cx = ((float)gx + 0.5f) * s, cy = ((float)gy + 0.5f) * s;
  float rs = 2.5f * s;
  int fg = 0;
  for (int g = 0; g < NG; ++g) {
    const float* m = meta + ((size_t)b * NG + g) * 6;
    if (m[0] == 0.f) continue;
    float bx = m[2], by = m[3], bw = m[4], bh = m[5];
    bool in_box = (cx > bx - 0.5f * bw) && (cx < bx + 0.5f * bw) &&
                  (cy > by - 0.5f * bh) && (cy < by + 0.5f * bh);
    bool in_ctr = (fabsf(cx - bx) < rs) && (fabsf(cy - by) < rs);
    if (in_box || in_ctr) { fg = 1; break; }
  }
  fgC[idx] = fg;
}

// ---------------------------------------------------------------- K4: WMMA onehot-GEMM cost matrix + fused IoU epilogue
// Each wave: one 16-anchor tile x TWO 16-gt tiles (g and g+16). A (Q rows) shared
// across both gt tiles; 2 accumulators, 6 WMMA issues. One-hot B built with one
// v_cmp + one 32-bit cndmask per dword.
__global__ void __launch_bounds__(256)
k4_cost_wmma(const _Float16* __restrict__ Q, const float* __restrict__ bbox,
             const float* __restrict__ sArr, const int* __restrict__ fgC,
             const float* __restrict__ meta, float* __restrict__ cost,
             float* __restrict__ iouA) {
  const int wave = threadIdx.x >> 5;
  const int lane = threadIdx.x & 31;
  const int atile = blockIdx.x * 8 + wave;
  if (atile >= ATILES) return;           // uniform per wave
  const int gpair = blockIdx.y;          // 0..1 : covers g in [p*32, p*32+32)
  const int b = blockIdx.z;
  const int hi = lane >> 4;              // lane[4]
  const int l16 = lane & 15;

  // per-lane GT columns: g0 = gpair*32 + l16, g1 = g0 + 16
  const int g0 = gpair * 32 + l16;
  const int g1 = g0 + 16;
  GtInfo gi0 = {0, 0.f, 0.f, 0.f, 0.f}, gi1 = {0, 0.f, 0.f, 0.f, 0.f};
  int cg0 = -1, cg1 = -1;
  {
    const float* m0 = meta + ((size_t)b * NG + g0) * 6;   // g0 < 48 always valid idx? g0<=47 ✓ (gpair<=1, l16<=15)
    gi0.valid = (m0[0] != 0.f);
    cg0 = (int)m0[1];
    gi0.bx = m0[2]; gi0.by = m0[3]; gi0.bw = m0[4]; gi0.bh = m0[5];
  }
  if (g1 < NG) {
    const float* m1 = meta + ((size_t)b * NG + g1) * 6;
    gi1.valid = (m1[0] != 0.f);
    cg1 = (int)m1[1];
    gi1.bx = m1[2]; gi1.by = m1[3]; gi1.bw = m1[4]; gi1.bh = m1[5];
  }

  // A-fragment anchor row (lanes L and L+16 share M = lane&15)
  const int aA = atile * 16 + l16;
  const _Float16* qrow = Q + ((size_t)b * NA + aA) * KPAD;

  // one-hot dword pattern: which 16-bit half of the dword holds 1.0h
  const int lb0 = cg0 - (hi << 4);             // local K index base (may be <0)
  const int lb1 = cg1 - (hi << 4);
  const unsigned one0 = (lb0 & 1) ? 0x3C000000u : 0x00003C00u;
  const unsigned one1 = (lb1 & 1) ? 0x3C000000u : 0x00003C00u;

  v8f acc0 = {}, acc1 = {};
  for (int kk = 0; kk < 3; ++kk) {
    union { v16h v; _Float16 h[16]; } A;
    union { v16h v; unsigned u[8]; } B0, B1;
    // A: halves 0..15 -> K = kk*32 + 16*hi + h
    const _Float16* qp = qrow + kk * 32 + hi * 16;
#pragma unroll
    for (int h = 0; h < 8; ++h) { A.h[h] = qp[h]; A.h[8 + h] = qp[8 + h]; }
    // B: onehot^T. dword d covers K = kk*32 + 16*hi + {2d, 2d+1}
    const unsigned s0 = (unsigned)(lb0 - kk * 32) >> 1;
    const unsigned s1 = (unsigned)(lb1 - kk * 32) >> 1;
#pragma unroll
    for (int d = 0; d < 8; ++d) {
      B0.u[d] = (s0 == (unsigned)d) ? one0 : 0u;
      B1.u[d] = (s1 == (unsigned)d) ? one1 : 0u;
    }
    acc0 = __builtin_amdgcn_wmma_f32_16x16x32_f16(false, A.v, false, B0.v,
                                                  (short)0, acc0, false, false);
    acc1 = __builtin_amdgcn_wmma_f32_16x16x32_f16(false, A.v, false, B1.v,
                                                  (short)0, acc1, false, false);
  }

  union { v8f v; float f[8]; } D0, D1;
  D0.v = acc0; D1.v = acc1;
#pragma unroll
  for (int r = 0; r < 8; ++r) {
    const int a = atile * 16 + r + hi * 8;           // M = r + 8*(lane>=16)
    const float* pb = bbox + ((size_t)b * NA + a) * 4;
    const float px = pb[0], py = pb[1], pw = pb[2], ph = pb[3];
    int gx, gy, lvl, hw, loc; float st;
    anchor_info(a, gx, gy, st, lvl, hw, loc);
    const float cx = ((float)gx + 0.5f) * st, cy = ((float)gy + 0.5f) * st;
    const float rs = 2.5f * st;
    const int   fgc = fgC[(size_t)b * NA + a];
    const float sA  = sArr[(size_t)b * NA + a];
    const float parea = pw * ph;

#pragma unroll
    for (int t = 0; t < 2; ++t) {
      const GtInfo& gi = t ? gi1 : gi0;
      const int g = t ? g1 : g0;
      if (g >= NG) continue;
      // pairwise IoU (gt, pred)
      float tlx = fmaxf(gi.bx - 0.5f * gi.bw, px - 0.5f * pw);
      float tly = fmaxf(gi.by - 0.5f * gi.bh, py - 0.5f * ph);
      float brx = fminf(gi.bx + 0.5f * gi.bw, px + 0.5f * pw);
      float bry = fminf(gi.by + 0.5f * gi.bh, py + 0.5f * ph);
      float en = (tlx < brx && tly < bry) ? 1.f : 0.f;
      float inter = (brx - tlx) * (bry - tly) * en;
      float uni = gi.bw * gi.bh + parea - inter + 1e-12f;
      float iou = inter / uni;
      bool in_box = (cx > gi.bx - 0.5f * gi.bw) && (cx < gi.bx + 0.5f * gi.bw) &&
                    (cy > gi.by - 0.5f * gi.bh) && (cy < gi.by + 0.5f * gi.bh);
      bool in_ctr = (fabsf(cx - gi.bx) < rs) && (fabsf(cy - gi.by) < rs);
      bool geom = in_box && in_ctr;
      bool cand = (fgc != 0) && gi.valid;
      float clsc = (t ? D1.f[r] : D0.f[r]) - sA;
      float c = clsc - 3.f * logf(iou + 1e-8f) +
                (geom ? 0.f : 100000.f) + (cand ? 0.f : 1000000000.f);
      size_t off = ((size_t)b * NG + g) * NA + a;
      cost[off] = c;
      iouA[off] = iou;
    }
  }
}

// ---------------------------------------------------------------- K5: per-GT dyn-k threshold (one wave / GT row)
__global__ void __launch_bounds__(256)
k5_thr(const float* __restrict__ cost, const float* __restrict__ iouA,
       const int* __restrict__ fgC, const float* __restrict__ meta,
       float* __restrict__ thr) {
  __shared__ float sm[8][32][20];
  const int wave = threadIdx.x >> 5, lane = threadIdx.x & 31;
  const int row = blockIdx.x * 8 + wave;     // 0 .. NB*NG-1 (exact: 200 blocks * 8)
  const int b = row / NG, g = row - b * NG;
  const int gv = (meta[((size_t)b * NG + g) * 6] != 0.f);
  float ti[10], tc[10];
#pragma unroll
  for (int j = 0; j < 10; ++j) { ti[j] = 0.f; tc[j] = 3.0e38f; }
  const float* crow = cost + (size_t)row * NA;
  const float* irow = iouA + (size_t)row * NA;
  const int*   fb   = fgC + (size_t)b * NA;
  for (int n = lane; n < NA; n += 32) {
    ins_asc(tc, crow[n]);
    float im = (gv && fb[n]) ? irow[n] : 0.f;
    ins_desc(ti, im);
  }
#pragma unroll
  for (int j = 0; j < 10; ++j) { sm[wave][lane][j] = ti[j]; sm[wave][lane][10 + j] = tc[j]; }
  __syncthreads();
  if (lane == 0) {
    float mi[10], mc[10];
#pragma unroll
    for (int j = 0; j < 10; ++j) { mi[j] = 0.f; mc[j] = 3.0e38f; }
    for (int l = 0; l < 32; ++l)
      for (int j = 0; j < 10; ++j) { ins_desc(mi, sm[wave][l][j]); ins_asc(mc, sm[wave][l][10 + j]); }
    float ssum = 0.f;
    for (int j = 0; j < 10; ++j) ssum += mi[j];
    int k = (int)ssum;                       // trunc, == astype(int32)
    k = (k < 1) ? 1 : (k > 10 ? 10 : k);     // dyn_k <= 10 since iou <= 1
    thr[row] = mc[k - 1];
  }
}

// ---------------------------------------------------------------- K6: match resolution per anchor
__global__ void k6_match(const float* __restrict__ cost, const float* __restrict__ iouA,
                         const int* __restrict__ fgC, const float* __restrict__ meta,
                         const float* __restrict__ thr,
                         int* __restrict__ fgF, int* __restrict__ mgt,
                         float* __restrict__ piou) {
  int idx = blockIdx.x * blockDim.x + threadIdx.x;
  if (idx >= NB * NA) return;
  int b = idx / NA, n = idx - b * NA;
  int fgc = fgC[idx];
  unsigned long long m = 0ull; int count = 0;
  float bestc = 3.4e38f; int bestg = 0;
  for (int g = 0; g < NG; ++g) {
    float c = cost[((size_t)b * NG + g) * NA + n];
    if (c < bestc) { bestc = c; bestg = g; }              // first-min tie break
    int gv = (meta[((size_t)b * NG + g) * 6] != 0.f);
    if (fgc && gv && c <= thr[b * NG + g]) { m |= (1ull << g); ++count; }
  }
  if (count > 1) m &= (1ull << bestg);                    // keep best GT only
  float pi = 0.f;
  unsigned long long mm = m;
  while (mm) {
    int g = __builtin_ctzll(mm);
    pi += iouA[((size_t)b * NG + g) * NA + n];
    mm &= mm - 1;
  }
  fgF[idx]  = (m != 0ull);
  mgt[idx]  = (m != 0ull) ? __builtin_ctzll(m) : 0;       // argmax(match) == first set
  piou[idx] = pi;
}

// ---------------------------------------------------------------- K7: losses, per-block partials
__global__ void __launch_bounds__(256)
k7_loss(const float* __restrict__ o0, const float* __restrict__ o1,
        const float* __restrict__ o2, const float* __restrict__ bbox,
        const float* __restrict__ objL, const float* __restrict__ meta,
        const int* __restrict__ fgF, const int* __restrict__ mgt,
        const float* __restrict__ piou, float* __restrict__ partials) {
  int idx = blockIdx.x * blockDim.x + threadIdx.x;
  float li = 0.f, lo = 0.f, lc = 0.f, nf = 0.f;
  if (idx < NB * NA) {
    int b = idx / NA, n = idx - b * NA;
    int fg = fgF[idx];
    float fgf = fg ? 1.f : 0.f;
    lo = bce_logit(objL[idx], fgf);
    nf = fgf;
    if (fg) {
      int g = mgt[idx];
      const float* m = meta + ((size_t)b * NG + g) * 6;
      int mc = (int)m[1];
      float tx = m[2], ty = m[3], tw = m[4], th = m[5];
      const float* pb = bbox + (size_t)idx * 4;
      float px = pb[0], py = pb[1], pw = pb[2], ph = pb[3];
      float tlx = fmaxf(px - 0.5f * pw, tx - 0.5f * tw);
      float tly = fmaxf(py - 0.5f * ph, ty - 0.5f * th);
      float brx = fminf(px + 0.5f * pw, tx + 0.5f * tw);
      float bry = fminf(py + 0.5f * ph, ty + 0.5f * th);
      float en = (tlx < brx && tly < bry) ? 1.f : 0.f;
      float inter = fmaxf(brx - tlx, 0.f) * fmaxf(bry - tly, 0.f) * en;
      float uni = pw * ph + tw * th - inter + 1e-16f;
      float iou = inter / uni;
      li = 1.f - iou * iou;
      float pi = piou[idx];
      for (int c = 0; c < NCLS; ++c) {
        float x = read_raw(o0, o1, o2, b, 5 + c, n);
        float y = (c == mc) ? pi : 0.f;
        lc += bce_logit(x, y);
      }
    }
  }
  __shared__ float4 sm[256];
  sm[threadIdx.x] = make_float4(li, lo, lc, nf);
  __syncthreads();
  for (int s = 128; s > 0; s >>= 1) {
    if (threadIdx.x < (unsigned)s) {
      float4 a = sm[threadIdx.x], b4 = sm[threadIdx.x + s];
      sm[threadIdx.x] = make_float4(a.x + b4.x, a.y + b4.y, a.z + b4.z, a.w + b4.w);
    }
    __syncthreads();
  }
  if (threadIdx.x == 0) {
    partials[(size_t)blockIdx.x * 4 + 0] = sm[0].x;
    partials[(size_t)blockIdx.x * 4 + 1] = sm[0].y;
    partials[(size_t)blockIdx.x * 4 + 2] = sm[0].z;
    partials[(size_t)blockIdx.x * 4 + 3] = sm[0].w;
  }
}

// ---------------------------------------------------------------- K8: final deterministic reduce
__global__ void __launch_bounds__(256)
k8_final(const float* __restrict__ partials, int nblk, float* __restrict__ out) {
  __shared__ float4 sm[256];
  float4 acc = make_float4(0.f, 0.f, 0.f, 0.f);
  for (int i = threadIdx.x; i < nblk; i += 256) {
    acc.x += partials[(size_t)i * 4 + 0];
    acc.y += partials[(size_t)i * 4 + 1];
    acc.z += partials[(size_t)i * 4 + 2];
    acc.w += partials[(size_t)i * 4 + 3];
  }
  sm[threadIdx.x] = acc;
  __syncthreads();
  for (int s = 128; s > 0; s >>= 1) {
    if (threadIdx.x < (unsigned)s) {
      float4 a = sm[threadIdx.x], b4 = sm[threadIdx.x + s];
      sm[threadIdx.x] = make_float4(a.x + b4.x, a.y + b4.y, a.z + b4.z, a.w + b4.w);
    }
    __syncthreads();
  }
  if (threadIdx.x == 0) {
    float numfg = fmaxf(sm[0].w, 1.f);
    out[0] = (5.f * sm[0].x + sm[0].y + sm[0].z) / numfg;
  }
}

// ---------------------------------------------------------------- launch
extern "C" void kernel_launch(void* const* d_in, const int* in_sizes, int n_in,
                              void* d_out, int out_size, void* d_ws, size_t ws_size,
                              hipStream_t stream) {
  const float* o0 = (const float*)d_in[0];   // (32,85,80,80)
  const float* o1 = (const float*)d_in[1];   // (32,85,40,40)
  const float* o2 = (const float*)d_in[2];   // (32,85,20,20)
  const float* labels = (const float*)d_in[3]; // (32,50,5)
  float* out = (float*)d_out;

  // workspace carve-out (~170 MB total)
  size_t off = 0;
  char* base = (char*)d_ws;
  auto take = [&](size_t bytes) -> char* {
    char* r = base + off;
    off += (bytes + 255) & ~(size_t)255;
    return r;
  };
  float*    bbox  = (float*)   take((size_t)NB * NA * 4 * sizeof(float));
  float*    objL  = (float*)   take((size_t)NB * NA * sizeof(float));
  float*    sArr  = (float*)   take((size_t)NB * NA * sizeof(float));
  int*      fgC   = (int*)     take((size_t)NB * NA * sizeof(int));
  float*    meta  = (float*)   take((size_t)NB * NG * 6 * sizeof(float));
  _Float16* Q     = (_Float16*)take((size_t)NB * NA * KPAD * sizeof(_Float16));
  float*    cost  = (float*)   take((size_t)NB * NG * NA * sizeof(float));
  float*    iouA  = (float*)   take((size_t)NB * NG * NA * sizeof(float));
  float*    thr   = (float*)   take((size_t)NB * NG * sizeof(float));
  int*      fgF   = (int*)     take((size_t)NB * NA * sizeof(int));
  int*      mgt   = (int*)     take((size_t)NB * NA * sizeof(int));
  float*    piou  = (float*)   take((size_t)NB * NA * sizeof(float));
  const int NBLK  = (NB * NA + 255) / 256;   // 1050
  float*    parts = (float*)   take((size_t)NBLK * 4 * sizeof(float));
  (void)ws_size; (void)in_sizes; (void)n_in; (void)out_size;

  k0_meta  <<<(NB * NG + 255) / 256, 256, 0, stream>>>(labels, meta);
  k1_decode<<<NBLK, 256, 0, stream>>>(o0, o1, o2, bbox, objL);
  k2_prob  <<<NBLK, 256, 0, stream>>>(o0, o1, o2, objL, Q, sArr);
  k3_fg    <<<NBLK, 256, 0, stream>>>(meta, fgC);
  k4_cost_wmma<<<dim3((ATILES + 7) / 8, 2, NB), 256, 0, stream>>>(Q, bbox, sArr, fgC,
                                                                  meta, cost, iouA);
  k5_thr   <<<(NB * NG) / 8, 256, 0, stream>>>(cost, iouA, fgC, meta, thr);
  k6_match <<<NBLK, 256, 0, stream>>>(cost, iouA, fgC, meta, thr, fgF, mgt, piou);
  k7_loss  <<<NBLK, 256, 0, stream>>>(o0, o1, o2, bbox, objL, meta, fgF, mgt, piou, parts);
  k8_final <<<1, 256, 0, stream>>>(parts, NBLK, out);
}
